// KBBlock_sss_29583734735356
// MI455X (gfx1250) — matrix operand order repaired and checked by reading, not compiled, not run on hardware
//
#include <hip/hip_runtime.h>

typedef __attribute__((ext_vector_type(16))) _Float16 v16h;
typedef __attribute__((ext_vector_type(8)))  _Float16 v8h;
typedef __attribute__((ext_vector_type(4)))  _Float16 v4h;
typedef __attribute__((ext_vector_type(8)))  float    v8f;

#define B_    2
#define C_    64
#define H_    160
#define W_    160
#define HW_   25600
#define P_    51200      // B_*HW_
#define NSET_ 32
#define KBN_  2368       // 2304 kb_w cols + 64 kb_b bias cols packed as extra N

// ---------------- WMMA fragment helpers (CDNA5 16x16x32 f16, wave32) -------

// A-matrix 16x32 f16 per ISA 7.12.2: lane M = l&15, hi = l>>4.
// Per lane the fragment is two contiguous 8xf16 runs: [k0+hi*8 .. +7] and
// [k0+hi*8+16 .. +23]  -> two b128 loads + register concat.
__device__ __forceinline__ v16h load_a16(const _Float16* src, int stride, int p0, int k0) {
  int lane = threadIdx.x & 31;
  int M = lane & 15, hi = lane >> 4;
  const _Float16* b = src + (size_t)(p0 + M) * stride + k0 + hi * 8;
  v8h x0 = *(const v8h*)(b);
  v8h x1 = *(const v8h*)(b + 16);
  return __builtin_shufflevector(x0, x1, 0,1,2,3,4,5,6,7,8,9,10,11,12,13,14,15);
}

// B fragments are pre-swizzled by k_prep into per-lane contiguous 16xf16
// records: frag[(fragIdx*32 + lane)*16 + e]  -> two coalesced b128 loads.
__device__ __forceinline__ v16h load_b_swz(const _Float16* frags, int fragIdx) {
  int lane = threadIdx.x & 31;
  const v8h* p = (const v8h*)(frags + ((size_t)fragIdx * 32 + lane) * 16);
  v8h x0 = p[0];
  v8h x1 = p[1];
  return __builtin_shufflevector(x0, x1, 0,1,2,3,4,5,6,7,8,9,10,11,12,13,14,15);
}

__device__ __forceinline__ v8f wmma32(v16h a, v16h b, v8f c) {
  return __builtin_amdgcn_wmma_f32_16x16x32_f16(false, a, false, b, (short)0, c, false, false);
}

// ---------------- K0: weight prep -> f16 fragments in WMMA lane order ------

// B 32x16 layout: lane N = n0+(l&15), hi=l>>4; element e: K = k0+hi*8+(e&7)+(e>>3)*16
template <typename F>
__device__ __forceinline__ void fill_frags(_Float16* dst, int Kdim, int Ndim, F get) {
  int ntiles = Ndim >> 4;
  int total = (Kdim >> 5) * ntiles * 512;
  for (int idx = threadIdx.x; idx < total; idx += blockDim.x) {
    int e    = idx & 15;
    int lane = (idx >> 4) & 31;
    int frag = idx >> 9;
    int kc = frag / ntiles, nt = frag % ntiles;
    int hi = lane >> 4, nl = lane & 15;
    int k = kc * 32 + hi * 8 + (e & 7) + (e >> 3) * 16;
    int n = nt * 16 + nl;
    dst[idx] = (_Float16)get(k, n);
  }
}

__global__ void k_prep(const float* kb_w, const float* kb_b,
                       const float* c11a_w, const float* conv3_w,
                       const float* conv4_w, const float* conv5_w,
                       _Float16* kbw_f, _Float16* c11a_f, _Float16* c3_f,
                       _Float16* c4_f, _Float16* c5_f) {
  fill_frags(kbw_f, 32, KBN_, [=](int k, int n) {
    return (n < 2304) ? kb_w[k * 2304 + n] : kb_b[k * 64 + (n - 2304)];
  });
  fill_frags(c11a_f, 64, 64,  [=](int k, int n) { return c11a_w[n * 64 + k]; });
  fill_frags(c3_f,   64, 64,  [=](int k, int n) { return conv3_w[n * 64 + k]; });
  fill_frags(c4_f,   64, 128, [=](int k, int n) { return conv4_w[n * 64 + k]; });
  fill_frags(c5_f,   64, 64,  [=](int k, int n) { return conv5_w[n * 64 + k]; });
}

// ---------------- K1: LayerNorm1 -> xh (f16 pixel-major) + channel partials

__global__ void k_ln1(const float* inp, const float* w, const float* bb,
                      _Float16* xh, float* partial) {
  __shared__ float wsum[8][C_];
  int p = blockIdx.x * 256 + threadIdx.x;
  int b = p / HW_, pix = p % HW_;
  const float* src = inp + (size_t)b * C_ * HW_ + pix;
  float v[C_];
  float mu = 0.f;
#pragma unroll
  for (int c = 0; c < C_; ++c) { v[c] = src[(size_t)c * HW_]; mu += v[c]; }
  mu *= (1.f / 64.f);
  float var = 0.f;
#pragma unroll
  for (int c = 0; c < C_; ++c) { float d = v[c] - mu; var += d * d; }
  float rinv = rsqrtf(var * (1.f / 64.f) + 1e-6f);
  int lane = threadIdx.x & 31, wv = threadIdx.x >> 5;
#pragma unroll
  for (int c = 0; c < C_; ++c) {
    float yv = w[c] * ((v[c] - mu) * rinv) + bb[c];
    v[c] = yv;
    float s = yv;
    s += __shfl_down(s, 16);
    s += __shfl_down(s, 8);
    s += __shfl_down(s, 4);
    s += __shfl_down(s, 2);
    s += __shfl_down(s, 1);
    if (lane == 0) wsum[wv][c] = s;
  }
  v8h* dst8 = (v8h*)(xh + (size_t)p * C_);
#pragma unroll
  for (int c8 = 0; c8 < 8; ++c8) {
    v8h pk;
#pragma unroll
    for (int e = 0; e < 8; ++e) pk[e] = (_Float16)v[c8 * 8 + e];
    dst8[c8] = pk;
  }
  __syncthreads();
  if (threadIdx.x < C_) {
    float s = 0.f;
#pragma unroll
    for (int k = 0; k < 8; ++k) s += wsum[k][threadIdx.x];
    partial[blockIdx.x * C_ + threadIdx.x] = s;
  }
}

// ---------------- K2: finish pooling + sca 1x1 conv (deterministic) --------

__global__ void k_sca(const float* partial, const float* sca_w, const float* sca_b,
                      float* sca) {
  __shared__ float pooled[B_ * C_];
  int t = threadIdx.x;           // 128 threads
  int b = t >> 6, c = t & 63;
  float s = 0.f;
  for (int k = 0; k < 100; ++k) s += partial[(b * 100 + k) * C_ + c];
  pooled[t] = s * (1.f / (float)HW_);
  __syncthreads();
  float acc = sca_b[c];
#pragma unroll
  for (int i = 0; i < 64; ++i) acc += sca_w[c * 64 + i] * pooled[b * 64 + i];
  sca[t] = acc;
}

// ---------------- K3: attention branch (3x3 grouped conv, gate, 1x1) -------

__global__ void k_att(const _Float16* xh, const float* c2a_w, const float* c2a_b,
                      const float* c2b_w, const float* c2b_b, const float* attg,
                      _Float16* att_h) {
  int p = blockIdx.x * 256 + threadIdx.x;
  int b = p / HW_, pix = p % HW_;
  int y = pix / W_, x = pix % W_;
  float t32[32];
#pragma unroll
  for (int o = 0; o < 32; ++o) t32[o] = c2a_b[o];
  for (int kh = 0; kh < 3; ++kh) {
    int yy = y + kh - 1;
    if (yy < 0 || yy >= H_) continue;
    for (int kw = 0; kw < 3; ++kw) {
      int xx = x + kw - 1;
      if (xx < 0 || xx >= W_) continue;
      const v4h* q = (const v4h*)(xh + (size_t)(b * HW_ + yy * W_ + xx) * C_);
      int tap = kh * 3 + kw;
#pragma unroll
      for (int j = 0; j < 16; ++j) {          // out ch 2j uses in 4j,4j+1; 2j+1 uses 4j+2,4j+3
        v4h qv = q[j];
        t32[2*j]   += (float)qv[0] * c2a_w[(4*j + 0) * 9 + tap]
                    + (float)qv[1] * c2a_w[(4*j + 1) * 9 + tap];
        t32[2*j+1] += (float)qv[2] * c2a_w[(4*j + 2) * 9 + tap]
                    + (float)qv[3] * c2a_w[(4*j + 3) * 9 + tap];
      }
    }
  }
  float g[16];
#pragma unroll
  for (int j = 0; j < 16; ++j) g[j] = t32[j] * t32[j + 16];
  float av[32];
#pragma unroll
  for (int s = 0; s < 32; ++s) {
    float a = c2b_b[s];
#pragma unroll
    for (int j = 0; j < 16; ++j) a += c2b_w[s * 16 + j] * g[j];
    av[s] = a * attg[s];
  }
  v8h* dst8 = (v8h*)(att_h + (size_t)p * NSET_);
#pragma unroll
  for (int s8 = 0; s8 < 4; ++s8) {
    v8h pk;
#pragma unroll
    for (int e = 0; e < 8; ++e) pk[e] = (_Float16)av[s8 * 8 + e];
    dst8[s8] = pk;
  }
}

// ---------------- K4a: 1x1 conv 64->64 via WMMA ----------------------------

__global__ void k_pw1(const _Float16* xh, const _Float16* c11a_f, const float* bias,
                      _Float16* pw_h) {
  int wave = threadIdx.x >> 5, lane = threadIdx.x & 31;
  int pix0 = blockIdx.x * 16;
  v16h a0 = load_a16(xh, C_, pix0, 0);
  v16h a1 = load_a16(xh, C_, pix0, 32);
  v16h b0 = load_b_swz(c11a_f, 0 * 4 + wave);
  v16h b1 = load_b_swz(c11a_f, 1 * 4 + wave);
  v8f c = {};
  c = wmma32(a0, b0, c);
  c = wmma32(a1, b1, c);
  int o = wave * 16 + (lane & 15), hi = lane >> 4;
  float bo = bias[o];
#pragma unroll
  for (int r = 0; r < 8; ++r) {
    int p = pix0 + r + 8 * hi;
    pw_h[(size_t)p * C_ + o] = (_Float16)(c[r] + bo);
  }
}

// ---------------- K4b: 5x5 grouped conv (groups=16, 4ch) -------------------

__global__ void k_gc5(const _Float16* pw_h, const float* w5, const float* b5,
                      _Float16* uf_h) {
  int p = blockIdx.x * 256 + threadIdx.x;
  int b = p / HW_, pix = p % HW_;
  int y = pix / W_, x = pix % W_;
  float acc[C_];
#pragma unroll
  for (int c = 0; c < C_; ++c) acc[c] = b5[c];
  for (int kh = 0; kh < 5; ++kh) {
    int yy = y + kh - 2;
    if (yy < 0 || yy >= H_) continue;
    for (int kw = 0; kw < 5; ++kw) {
      int xx = x + kw - 2;
      if (xx < 0 || xx >= W_) continue;
      const v4h* q = (const v4h*)(pw_h + (size_t)(b * HW_ + yy * W_ + xx) * C_);
      int tap = kh * 5 + kw;
#pragma unroll
      for (int gi = 0; gi < 16; ++gi) {
        v4h qv = q[gi];
        float q0 = qv[0], q1 = qv[1], q2 = qv[2], q3 = qv[3];
#pragma unroll
        for (int oc = 0; oc < 4; ++oc) {
          int c = gi * 4 + oc;
          acc[c] += q0 * w5[(c * 4 + 0) * 25 + tap] + q1 * w5[(c * 4 + 1) * 25 + tap]
                  + q2 * w5[(c * 4 + 2) * 25 + tap] + q3 * w5[(c * 4 + 3) * 25 + tap];
        }
      }
    }
  }
  v4h* dst = (v4h*)(uf_h + (size_t)p * C_);
#pragma unroll
  for (int gi = 0; gi < 16; ++gi) {
    v4h o4;
#pragma unroll
    for (int e = 0; e < 4; ++e) o4[e] = (_Float16)acc[gi * 4 + e];
    dst[gi] = o4;
  }
}

// ---------------- K5: fused KBA -- WMMA weight-gen into LDS + einsum -------
// per block: 16 pixels, 8 waves. attk tile [16 x 2368] f32 lives only in LDS.

__global__ void __launch_bounds__(256) k_kba(const _Float16* att_h, const _Float16* kbw_f,
                      const _Float16* uf_h, const float* ga1, const float* sca,
                      _Float16* z_h) {
  extern __shared__ float lds[];   // 16 * KBN_ floats = 148 KB
  int wave = threadIdx.x >> 5, lane = threadIdx.x & 31;
  int hi = lane >> 4, nl = lane & 15;
  int pix0 = blockIdx.x * 16;

  // phase 1: attk[16 x 2368] = att[16 x 32] @ [kb_w | kb_b]  (one WMMA/tile)
  v16h a = load_a16(att_h, NSET_, pix0, 0);
  for (int t = wave; t < KBN_ / 16; t += 8) {
    __builtin_prefetch(kbw_f + ((size_t)(t + 8) * 32 + lane) * 16, 0, 1);
    v16h b = load_b_swz(kbw_f, t);
    v8f d = {};
    d = wmma32(a, b, d);
    int n0 = t * 16;
#pragma unroll
    for (int r = 0; r < 8; ++r)
      lds[(r + 8 * hi) * KBN_ + n0 + nl] = d[r];
  }
  __syncthreads();

  // phase 2: per-pixel grouped einsum. thread = (pixel pl, group g)
  int pl = threadIdx.x >> 4, g = threadIdx.x & 15;
  int p = pix0 + pl;
  int b = p / HW_, pix = p % HW_;
  int y = pix / W_, x = pix % W_;
  float pt[36];
#pragma unroll
  for (int kh = 0; kh < 3; ++kh) {
#pragma unroll
    for (int kw = 0; kw < 3; ++kw) {
      int yy = y + kh - 1, xx = x + kw - 1;
      int tap = kh * 3 + kw;
      v4h qv = {};
      if (yy >= 0 && yy < H_ && xx >= 0 && xx < W_)
        qv = *(const v4h*)(uf_h + (size_t)(b * HW_ + yy * W_ + xx) * C_ + g * 4);
#pragma unroll
      for (int ic = 0; ic < 4; ++ic) pt[ic * 9 + tap] = (float)qv[ic];
    }
  }
  const float* row = lds + pl * KBN_;
  v4h ufv = *(const v4h*)(uf_h + (size_t)p * C_ + g * 4);
  v4h z4;
#pragma unroll
  for (int o = 0; o < 4; ++o) {
    int c = g * 4 + o;
    float acc = row[2304 + c];                 // packed kb_b bias
    const float4* wr = (const float4*)(row + g * 144 + o * 36);
#pragma unroll
    for (int i = 0; i < 9; ++i) {
      float4 w4 = wr[i];
      acc += w4.x * pt[4*i] + w4.y * pt[4*i+1] + w4.z * pt[4*i+2] + w4.w * pt[4*i+3];
    }
    z4[o] = (_Float16)((acc * ga1[c] + (float)ufv[o]) * sca[b * C_ + c]);
  }
  *(v4h*)(z_h + (size_t)p * C_ + g * 4) = z4;
}

// ---------------- K6: conv3 (WMMA) + residual -> y --------------------------

__global__ void k_c3(const _Float16* z_h, const _Float16* c3_f, const float* c3b,
                     const float* inp, const float* beta, float* y_f) {
  int wave = threadIdx.x >> 5, lane = threadIdx.x & 31;
  int pix0 = blockIdx.x * 16;
  v16h a0 = load_a16(z_h, C_, pix0, 0);
  v16h a1 = load_a16(z_h, C_, pix0, 32);
  v16h b0 = load_b_swz(c3_f, 0 * 4 + wave);
  v16h b1 = load_b_swz(c3_f, 1 * 4 + wave);
  v8f c = {};
  c = wmma32(a0, b0, c);
  c = wmma32(a1, b1, c);
  int o = wave * 16 + (lane & 15), hi = lane >> 4;
  float bo = c3b[o], be = beta[o];
#pragma unroll
  for (int r = 0; r < 8; ++r) {
    int p = pix0 + r + 8 * hi;
    int b = p / HW_, pix = p % HW_;
    float yv = inp[((size_t)b * C_ + o) * HW_ + pix] + be * (c[r] + bo);
    y_f[(size_t)p * C_ + o] = yv;
  }
}

// ---------------- K7a: LayerNorm2 -> f16 A-matrix ---------------------------

__global__ void k_ln2(const float* y_f, const float* w, const float* bb, _Float16* t_h) {
  int p = blockIdx.x * 256 + threadIdx.x;
  const float* src = y_f + (size_t)p * C_;
  float v[C_], mu = 0.f;
#pragma unroll
  for (int c = 0; c < C_; ++c) { v[c] = src[c]; mu += v[c]; }
  mu *= (1.f / 64.f);
  float var = 0.f;
#pragma unroll
  for (int c = 0; c < C_; ++c) { float d = v[c] - mu; var += d * d; }
  float rinv = rsqrtf(var * (1.f / 64.f) + 1e-6f);
  v8h* dst8 = (v8h*)(t_h + (size_t)p * C_);
#pragma unroll
  for (int c8 = 0; c8 < 8; ++c8) {
    v8h pk;
#pragma unroll
    for (int e = 0; e < 8; ++e) {
      int c = c8 * 8 + e;
      pk[e] = (_Float16)(w[c] * ((v[c] - mu) * rinv) + bb[c]);
    }
    dst8[c8] = pk;
  }
}

// ---------------- K7b: FFN fused (WMMA 64->128, gate, WMMA 64->64) ----------

__global__ void __launch_bounds__(256) k_ffn(const _Float16* t_h, const _Float16* c4_f, const float* c4b,
                      const _Float16* c5_f, const float* c5b,
                      const float* y_f, const float* gamma, float* out) {
  extern __shared__ float sm[];                  // t2[16*128] f32, gh[16*64] f16
  float* t2 = sm;
  _Float16* gh = (_Float16*)(sm + 16 * 128);
  int wave = threadIdx.x >> 5, lane = threadIdx.x & 31;
  int hi = lane >> 4, nl = lane & 15;
  int pix0 = blockIdx.x * 16;
  {
    int n0 = wave * 16;                          // 8 waves x 16 = 128 outputs
    v16h a0 = load_a16(t_h, C_, pix0, 0);
    v16h a1 = load_a16(t_h, C_, pix0, 32);
    v16h b0 = load_b_swz(c4_f, 0 * 8 + wave);
    v16h b1 = load_b_swz(c4_f, 1 * 8 + wave);
    v8f c = {};
    c = wmma32(a0, b0, c);
    c = wmma32(a1, b1, c);
    int o = n0 + nl;
    float bo = c4b[o];
#pragma unroll
    for (int r = 0; r < 8; ++r)
      t2[(r + 8 * hi) * 128 + o] = c[r] + bo;
  }
  __syncthreads();
#pragma unroll
  for (int k = 0; k < 4; ++k) {
    int idx = threadIdx.x + k * 256;
    int pl = idx >> 6, j = idx & 63;
    gh[pl * 64 + j] = (_Float16)(t2[pl * 128 + j] * t2[pl * 128 + 64 + j]);
  }
  __syncthreads();
  if (wave < 4) {                                // uniform per wave: EXEC all-1
    int n0 = wave * 16;
    v16h a0 = load_a16(gh, C_, 0, 0);
    v16h a1 = load_a16(gh, C_, 0, 32);
    v16h b0 = load_b_swz(c5_f, 0 * 4 + wave);
    v16h b1 = load_b_swz(c5_f, 1 * 4 + wave);
    v8f c = {};
    c = wmma32(a0, b0, c);
    c = wmma32(a1, b1, c);
    int o = n0 + nl;
    float bo = c5b[o], gm = gamma[o];
#pragma unroll
    for (int r = 0; r < 8; ++r) {
      int p = pix0 + r + 8 * hi;
      int b = p / HW_, pix = p % HW_;
      out[((size_t)b * C_ + o) * HW_ + pix] = y_f[(size_t)p * C_ + o] + gm * (c[r] + bo);
    }
  }
}

// ---------------- host launch ----------------------------------------------

extern "C" void kernel_launch(void* const* d_in, const int* in_sizes, int n_in,
                              void* d_out, int out_size, void* d_ws, size_t ws_size,
                              hipStream_t stream) {
  const float* inp     = (const float*)d_in[0];
  const float* ln1_w   = (const float*)d_in[1];
  const float* ln1_b   = (const float*)d_in[2];
  const float* ln2_w   = (const float*)d_in[3];
  const float* ln2_b   = (const float*)d_in[4];
  const float* sca_w   = (const float*)d_in[5];
  const float* sca_b   = (const float*)d_in[6];
  const float* c11a_w  = (const float*)d_in[7];
  const float* c11a_b  = (const float*)d_in[8];
  const float* c11b_w  = (const float*)d_in[9];
  const float* c11b_b  = (const float*)d_in[10];
  const float* c2a_w   = (const float*)d_in[11];
  const float* c2a_b   = (const float*)d_in[12];
  const float* c2b_w   = (const float*)d_in[13];
  const float* c2b_b   = (const float*)d_in[14];
  const float* conv3_w = (const float*)d_in[15];
  const float* conv3_b = (const float*)d_in[16];
  const float* conv4_w = (const float*)d_in[17];
  const float* conv4_b = (const float*)d_in[18];
  const float* conv5_w = (const float*)d_in[19];
  const float* conv5_b = (const float*)d_in[20];
  const float* kb_w    = (const float*)d_in[21];
  const float* kb_b    = (const float*)d_in[22];
  const float* ga1     = (const float*)d_in[23];
  const float* attg    = (const float*)d_in[24];
  const float* beta    = (const float*)d_in[25];
  const float* gamma   = (const float*)d_in[26];
  float* out = (float*)d_out;
  (void)in_sizes; (void)n_in; (void)out_size; (void)ws_size;

  char* ws = (char*)d_ws;
  size_t off = 0;
  auto take = [&](size_t bytes) -> void* {
    void* p = ws + off;
    off += (bytes + 255) & ~(size_t)255;
    return p;
  };
  _Float16* xh    = (_Float16*)take((size_t)P_ * C_ * 2);
  _Float16* att_h = (_Float16*)take((size_t)P_ * NSET_ * 2);
  _Float16* pw_h  = (_Float16*)take((size_t)P_ * C_ * 2);
  _Float16* uf_h  = (_Float16*)take((size_t)P_ * C_ * 2);
  _Float16* z_h   = (_Float16*)take((size_t)P_ * C_ * 2);
  _Float16* t_h   = (_Float16*)take((size_t)P_ * C_ * 2);
  float*    y_f   = (float*)take((size_t)P_ * C_ * 4);
  float*    part  = (float*)take((size_t)200 * C_ * 4);
  float*    sca   = (float*)take((size_t)B_ * C_ * 4);
  _Float16* kbw_f = (_Float16*)take((size_t)NSET_ * KBN_ * 2 + 4096); // + prefetch slop
  _Float16* c11a_f= (_Float16*)take((size_t)64 * 64 * 2);
  _Float16* c3_f  = (_Float16*)take((size_t)64 * 64 * 2);
  _Float16* c4_f  = (_Float16*)take((size_t)64 * 128 * 2);
  _Float16* c5_f  = (_Float16*)take((size_t)64 * 64 * 2);

  hipLaunchKernelGGL(k_prep, dim3(1), dim3(256), 0, stream,
                     kb_w, kb_b, c11a_w, conv3_w, conv4_w, conv5_w,
                     kbw_f, c11a_f, c3_f, c4_f, c5_f);
  hipLaunchKernelGGL(k_ln1, dim3(200), dim3(256), 0, stream,
                     inp, ln1_w, ln1_b, xh, part);
  hipLaunchKernelGGL(k_sca, dim3(1), dim3(128), 0, stream,
                     part, sca_w, sca_b, sca);
  hipLaunchKernelGGL(k_att, dim3(200), dim3(256), 0, stream,
                     xh, c2a_w, c2a_b, c2b_w, c2b_b, attg, att_h);
  hipLaunchKernelGGL(k_pw1, dim3(P_ / 16), dim3(128), 0, stream,
                     xh, c11a_f, c11a_b, pw_h);
  hipLaunchKernelGGL(k_gc5, dim3(200), dim3(256), 0, stream,
                     pw_h, c11b_w, c11b_b, uf_h);
  hipLaunchKernelGGL(k_kba, dim3(P_ / 16), dim3(256), (size_t)16 * KBN_ * 4, stream,
                     att_h, kbw_f, uf_h, ga1, sca, z_h);
  hipLaunchKernelGGL(k_c3, dim3(P_ / 16), dim3(128), 0, stream,
                     z_h, c3_f, conv3_b, inp, beta, y_f);
  hipLaunchKernelGGL(k_ln2, dim3(200), dim3(256), 0, stream,
                     y_f, ln2_w, ln2_b, t_h);
  hipLaunchKernelGGL(k_ffn, dim3(P_ / 16), dim3(256), (size_t)(16 * 128 * 4 + 16 * 64 * 2), stream,
                     t_h, c4_f, conv4_b, c5_f, conv5_b, y_f, gamma, out);
}
